// LINKX_86947317941135
// MI455X (gfx1250) — compile-verified, matrix operand order checked by reading, not run on hardware
//
#include <hip/hip_runtime.h>

// ---------------- problem constants ----------------
constexpr int N_ = 100000;   // nodes
constexpr int E_ = 1600000;  // edges
constexpr int D_ = 128;      // hidden dim
constexpr int C_ = 64;       // classes

typedef __attribute__((ext_vector_type(16))) __bf16 v16bf;
typedef __attribute__((ext_vector_type(8)))  float  v8f;

// slot s (0..15) of a 16-bit A/B fragment -> k within the 32-wide K tile
// (per CDNA5 ISA 16-bit A 16x32 layout; half = lane>>4)
__device__ __forceinline__ int kslot(int s, int half) {
  return ((s < 8) ? s : s + 8) + 8 * half;
}

// Load an A-fragment (16x32 bf16) from a row-major fp32 matrix.
// p must point at row m, column (kt*32 + 8*half).
__device__ __forceinline__ v16bf a_from_f32(const float* p) {
  v16bf a;
#pragma unroll
  for (int j = 0; j < 4; ++j) {
    float2 f0 = *(const float2*)(p + 2 * j);        // k = +2j, +2j+1
    float2 f1 = *(const float2*)(p + 16 + 2 * j);   // k = +16+2j, +17+2j
    a[2 * j]     = (__bf16)f0.x;
    a[2 * j + 1] = (__bf16)f0.y;
    a[8 + 2 * j]     = (__bf16)f1.x;
    a[8 + 2 * j + 1] = (__bf16)f1.y;
  }
  return a;
}

// Load an A-fragment from a row-major bf16 matrix (global or LDS).
__device__ __forceinline__ v16bf a_from_bf16(const __bf16* p) {
  union { v16bf v; unsigned u[8]; } r;
#pragma unroll
  for (int j = 0; j < 4; ++j) {
    r.u[j]     = *(const unsigned*)(p + 2 * j);
    r.u[4 + j] = *(const unsigned*)(p + 16 + 2 * j);
  }
  return r.v;
}

// Load a pre-packed B-fragment: blob is frag-major, 512 bf16 per fragment,
// lane-major (32 lanes x 16 elements, 32B contiguous per lane).
__device__ __forceinline__ v16bf b_frag(const __bf16* blob, int frag, int lane) {
  union { v16bf v; uint4 q[2]; } r;
  const uint4* p = (const uint4*)(blob + frag * 512 + lane * 16);
  r.q[0] = p[0];
  r.q[1] = p[1];
  return r.v;
}

__device__ __forceinline__ v8f wmma_bf16(v16bf a, v16bf b, v8f c) {
  return __builtin_amdgcn_wmma_f32_16x16x32_bf16(false, a, false, b,
                                                 (short)0, c, false, false);
}

// ---------------- kernels ----------------

__global__ void zero_f32(float* __restrict__ p, size_t n) {
  size_t i = (size_t)blockIdx.x * blockDim.x + threadIdx.x;
  size_t stride = (size_t)gridDim.x * blockDim.x;
  for (; i < n; i += stride) p[i] = 0.f;
}

// Pack a PyTorch-layout weight Wsrc [Nout, Kin] (row-major fp32) into bf16
// B-fragment blobs for B = Wsrc^T ([Kin x Nout]); frag = nt*(Kin/32) + kt.
__global__ void pack_bmat(const float* __restrict__ Wsrc, __bf16* __restrict__ blob,
                          int Nout, int Kin) {
  int KT = Kin / 32;
  int total = (Nout / 16) * KT * 512;
  for (int idx = blockIdx.x * blockDim.x + threadIdx.x; idx < total;
       idx += gridDim.x * blockDim.x) {
    int frag = idx >> 9, r = idx & 511;
    int L = r >> 4, s = r & 15;
    int nt = frag / KT, kt = frag % KT;
    int n = nt * 16 + (L & 15);
    int k = kt * 32 + kslot(s, L >> 4);
    blob[idx] = (__bf16)Wsrc[(size_t)n * Kin + k];
  }
}

// xA[row[e], :] += WA[:, col[e]]   (WA is [D, N] row-major; column gather
// hits L2 since WA (51.2MB) is L2-resident on the 192MB L2)
__global__ __launch_bounds__(256) void edge_scatter(
    const long long* __restrict__ row, const long long* __restrict__ col,
    const float* __restrict__ WA, float* __restrict__ xA) {
  int e = blockIdx.x * 2 + (threadIdx.x >> 7);
  int d = threadIdx.x & 127;
  if (e < E_) {
    int r = (int)row[e];
    int c = (int)col[e];
    atomicAdd(&xA[(size_t)r * D_ + d], WA[(size_t)d * N_ + c]);
  }
}

// Fused: xX = x@WX^T + bX ; h = relu([xA|xX]@W^T + bW + xA + xX) ;
// h1 = relu(h@W1^T + b1) -> H1 (bf16) ; accumulate BN sum/sumsq.
__global__ __launch_bounds__(256) void link_main(
    const float* __restrict__ x, const float* __restrict__ xA,
    const float* __restrict__ bA, const float* __restrict__ bX,
    const float* __restrict__ bW, const float* __restrict__ b1,
    const __bf16* __restrict__ blobWX, const __bf16* __restrict__ blobW,
    const __bf16* __restrict__ blobW1, __bf16* __restrict__ H1,
    float* __restrict__ gsum, float* __restrict__ gsq) {
  __shared__ __bf16 cat[8][16][256];  // 64KB: per-wave [xA | xX] staging
  const int lane = threadIdx.x & 31;
  const int wave = threadIdx.x >> 5;
  const int mbase = (blockIdx.x * 8 + wave) * 16;
  const int half = lane >> 4, nn = lane & 15;
  __bf16 (*mycat)[256] = cat[wave];

  // stage 0: xA(+bA) -> cat[:, 0:128] (bf16), rows clamped for tail tile
  for (int idx = lane; idx < 16 * 128; idx += 32) {
    int r = idx >> 7, c = idx & 127;
    int m = mbase + r; if (m >= N_) m = N_ - 1;
    mycat[r][c] = (__bf16)(xA[(size_t)m * D_ + c] + bA[c]);
  }

  int mrow = mbase + nn; if (mrow >= N_) mrow = N_ - 1;

  // stage 1: xX = x @ WX^T + bX  -> cat[:, 128:256]
  v16bf axf[4];
#pragma unroll
  for (int kt = 0; kt < 4; ++kt)
    axf[kt] = a_from_f32(x + (size_t)mrow * D_ + kt * 32 + half * 8);
#pragma unroll
  for (int nt = 0; nt < 8; ++nt) {
    v8f acc = {};
#pragma unroll
    for (int kt = 0; kt < 4; ++kt)
      acc = wmma_bf16(axf[kt], b_frag(blobWX, nt * 4 + kt, lane), acc);
    float bias = bX[nt * 16 + nn];
#pragma unroll
    for (int v = 0; v < 8; ++v)
      mycat[v + 8 * half][128 + nt * 16 + nn] = (__bf16)(acc[v] + bias);
  }
  __syncthreads();

  // stage 2: h = relu([xA|xX] @ W^T + bW + xA + xX), K = 256
  v16bf ac[8];
#pragma unroll
  for (int kt = 0; kt < 8; ++kt)
    ac[kt] = a_from_bf16(&mycat[nn][kt * 32 + half * 8]);
#pragma unroll
  for (int nt = 0; nt < 8; ++nt) {
    v8f acc = {};
#pragma unroll
    for (int kt = 0; kt < 8; ++kt)
      acc = wmma_bf16(ac[kt], b_frag(blobW, nt * 8 + kt, lane), acc);
    float bias = bW[nt * 16 + nn];
#pragma unroll
    for (int v = 0; v < 8; ++v) {
      int rr = v + 8 * half, colc = nt * 16 + nn;
      float res = (float)mycat[rr][colc] + (float)mycat[rr][128 + colc];
      float h = acc[v] + bias + res;
      mycat[rr][colc] = (__bf16)(h > 0.f ? h : 0.f);  // overwrite xA slot
    }
  }
  __syncthreads();

  // stage 3: h1 = relu(h @ W1^T + b1) -> H1; local BN partial sums
  v16bf ah[4];
#pragma unroll
  for (int kt = 0; kt < 4; ++kt)
    ah[kt] = a_from_bf16(&mycat[nn][kt * 32 + half * 8]);
  float ls[8], lsq[8];
#pragma unroll
  for (int nt = 0; nt < 8; ++nt) {
    v8f acc = {};
#pragma unroll
    for (int kt = 0; kt < 4; ++kt)
      acc = wmma_bf16(ah[kt], b_frag(blobW1, nt * 4 + kt, lane), acc);
    int colc = nt * 16 + nn;
    float bias = b1[colc];
    float s = 0.f, sq = 0.f;
#pragma unroll
    for (int v = 0; v < 8; ++v) {
      int m = mbase + v + 8 * half;
      float h1 = acc[v] + bias;
      h1 = h1 > 0.f ? h1 : 0.f;
      if (m < N_) {
        H1[(size_t)m * D_ + colc] = (__bf16)h1;
        s += h1;
        sq += h1 * h1;
      }
    }
    ls[nt] = s; lsq[nt] = sq;
  }
  __syncthreads();

  // block-level BN reduction in (now free) LDS, then 128 global atomics
  float* bn = reinterpret_cast<float*>(&cat[0][0][0]);  // [0:128]=sum,[128:256]=sq
  if (threadIdx.x < 256) bn[threadIdx.x] = 0.f;
  __syncthreads();
#pragma unroll
  for (int nt = 0; nt < 8; ++nt) {
    int colc = nt * 16 + nn;
    atomicAdd(&bn[colc], ls[nt]);
    atomicAdd(&bn[128 + colc], lsq[nt]);
  }
  __syncthreads();
  if (threadIdx.x < 128) {
    atomicAdd(&gsum[threadIdx.x], bn[threadIdx.x]);
    atomicAdd(&gsq[threadIdx.x], bn[128 + threadIdx.x]);
  }
}

// Fold BN affine into W2: out = h1 @ (diag(a) W2^T) + (c @ W2^T + b2)
__global__ __launch_bounds__(128) void bn_fold(
    const float* __restrict__ gsum, const float* __restrict__ gsq,
    const float* __restrict__ g1, const float* __restrict__ beta1,
    const float* __restrict__ W2, const float* __restrict__ b2,
    float* __restrict__ b2m, __bf16* __restrict__ blobW2m) {
  __shared__ float sa[128], sc[128];
  int t = threadIdx.x;
  float mu = gsum[t] * (1.0f / N_);
  float var = gsq[t] * (1.0f / N_) - mu * mu;
  var = var < 0.f ? 0.f : var;
  float a = g1[t] * rsqrtf(var + 1e-5f);
  sa[t] = a;
  sc[t] = beta1[t] - mu * a;
  __syncthreads();
  if (t < C_) {
    float s = b2[t];
    for (int k = 0; k < D_; ++k) s += sc[k] * W2[(size_t)t * D_ + k];
    b2m[t] = s;
  }
  // pack scaled W2 into B-fragments: KT=4, NT=4 -> 16 frags * 512 elems
  for (int idx = t; idx < 16 * 512; idx += 128) {
    int frag = idx >> 9, r = idx & 511;
    int L = r >> 4, s_ = r & 15;
    int nt = frag >> 2, kt = frag & 3;
    int n = nt * 16 + (L & 15);
    int k = kt * 32 + kslot(s_, L >> 4);
    blobW2m[idx] = (__bf16)(sa[k] * W2[(size_t)n * D_ + k]);
  }
}

// out = H1 @ W2m^T + b2m   ([N,128] x [128,64])
__global__ __launch_bounds__(256) void link_out(
    const __bf16* __restrict__ H1, const __bf16* __restrict__ blobW2m,
    const float* __restrict__ b2m, float* __restrict__ out) {
  const int lane = threadIdx.x & 31;
  const int wave = threadIdx.x >> 5;
  const int mbase = (blockIdx.x * 8 + wave) * 16;
  const int half = lane >> 4, nn = lane & 15;
  int mrow = mbase + nn; if (mrow >= N_) mrow = N_ - 1;
  v16bf ah[4];
#pragma unroll
  for (int kt = 0; kt < 4; ++kt)
    ah[kt] = a_from_bf16(H1 + (size_t)mrow * D_ + kt * 32 + half * 8);
#pragma unroll
  for (int nt = 0; nt < 4; ++nt) {
    v8f acc = {};
#pragma unroll
    for (int kt = 0; kt < 4; ++kt)
      acc = wmma_bf16(ah[kt], b_frag(blobW2m, nt * 4 + kt, lane), acc);
    int colc = nt * 16 + nn;
    float bias = b2m[colc];
#pragma unroll
    for (int v = 0; v < 8; ++v) {
      int m = mbase + v + 8 * half;
      if (m < N_) out[(size_t)m * C_ + colc] = acc[v] + bias;
    }
  }
}

// ---------------- host launcher ----------------
extern "C" void kernel_launch(void* const* d_in, const int* in_sizes, int n_in,
                              void* d_out, int out_size, void* d_ws, size_t ws_size,
                              hipStream_t stream) {
  (void)in_sizes; (void)n_in; (void)out_size; (void)ws_size;
  const float*     x   = (const float*)d_in[0];
  const long long* row = (const long long*)d_in[1];
  const long long* col = (const long long*)d_in[2];
  const float*     WA  = (const float*)d_in[3];
  const float*     bA  = (const float*)d_in[4];
  const float*     WX  = (const float*)d_in[5];
  const float*     bX  = (const float*)d_in[6];
  const float*     W   = (const float*)d_in[7];
  const float*     bW  = (const float*)d_in[8];
  const float*     W1  = (const float*)d_in[9];
  const float*     b1  = (const float*)d_in[10];
  const float*     g1  = (const float*)d_in[11];
  const float*     be1 = (const float*)d_in[12];
  const float*     W2  = (const float*)d_in[13];
  const float*     b2  = (const float*)d_in[14];
  float* out = (float*)d_out;

  // workspace layout (~73.4 MB total)
  float*  xA      = (float*)d_ws;                 // N*D fp32
  float*  gsum    = xA + (size_t)N_ * D_;         // 128
  float*  gsq     = gsum + D_;                    // 128
  float*  b2m     = gsq + D_;                     // 64
  __bf16* H1      = (__bf16*)(b2m + C_);          // N*D bf16
  __bf16* blobWX  = H1 + (size_t)N_ * D_;         // 32 frags  (K=128,N=128)
  __bf16* blobW   = blobWX + 32 * 512;            // 64 frags  (K=256,N=128)
  __bf16* blobW1  = blobW + 64 * 512;             // 32 frags  (K=128,N=128)
  __bf16* blobW2m = blobW1 + 32 * 512;            // 16 frags  (K=128,N=64)

  zero_f32<<<2048, 256, 0, stream>>>(xA, (size_t)N_ * D_ + 256);
  pack_bmat<<<64, 256, 0, stream>>>(WX, blobWX, 128, 128);
  pack_bmat<<<128, 256, 0, stream>>>(W, blobW, 128, 256);
  pack_bmat<<<64, 256, 0, stream>>>(W1, blobW1, 128, 128);
  edge_scatter<<<E_ / 2, 256, 0, stream>>>(row, col, WA, xA);
  link_main<<<(N_ + 127) / 128, 256, 0, stream>>>(
      x, xA, bA, bX, bW, b1, blobWX, blobW, blobW1, H1, gsum, gsq);
  bn_fold<<<1, 128, 0, stream>>>(gsum, gsq, g1, be1, W2, b2, b2m, blobW2m);
  link_out<<<(N_ + 127) / 128, 256, 0, stream>>>(H1, blobW2m, b2m, out);
}
